// FlowNetC_49426483642933
// MI455X (gfx1250) — matrix operand order repaired and checked by reading, not compiled, not run on hardware
//
#include <hip/hip_runtime.h>
#include <cstdint>

// ---------------------------------------------------------------------------
// FlowNetC forward for MI455X (gfx1250): bf16 WMMA implicit-GEMM convs +
// VALU correlation. 128x128x64 block tile, K-templated division-free staging,
// swizzled LDS, async global->LDS staging of the weight tile.
// ---------------------------------------------------------------------------

typedef __attribute__((ext_vector_type(16))) __bf16 v16bf;
typedef __attribute__((ext_vector_type(8)))  __bf16 v8bf;
typedef __attribute__((ext_vector_type(8)))  float  v8f;
typedef __attribute__((ext_vector_type(4)))  int    v4i;

#define BM 128
#define BN 128
#define BK 64
#define EPSV 1e-5f

#if __has_builtin(__builtin_amdgcn_global_load_async_to_lds_b128)
#define HAVE_ASYNC_LDS 1
#define AS3 __attribute__((address_space(3)))
#endif

struct ConvParams {
    const __bf16* X;    // [N, Ci, Hi, Wi] bf16
    const __bf16* Wt;   // [Copad, Kpad]   bf16, zero padded
    const float *gamma, *beta, *mean, *var;
    float*  outF;       // optional fp32 output [N, Cs, Ho, Wo] at channel off co_off
    __bf16* outB;       // optional bf16 output, same layout
    int N, Ci, Hi, Wi, Co, Ho, Wo, S, P;
    int Kpad, co_off, Cs;
};

// 16B-block XOR swizzle inside a 64-element (128B) row
__device__ __forceinline__ int swz(int row, int colElem) {
    return row * BK + ((((colElem >> 3) ^ (row & 7)) << 3) | (colElem & 7));
}

// ------------------------- implicit-GEMM conv + BN + ReLU -------------------
// KF = filter size (7,5,3,1), compile-time so the k->(ci,kh,kw) decomposition
// strength-reduces to magic-number multiplies and constant compares.
// 8 waves: 4 along M x 2 along N; wave tile 32x64 = 2x4 f32 accumulators.
template <int KF>
__global__ __launch_bounds__(256)
void conv_bn_relu_wmma(ConvParams p)
{
    __shared__ __align__(16) __bf16 lA[BM * BK];   // im2col tile  [m][k] (swizzled)
    __shared__ __align__(16) __bf16 lB[BN * BK];   // weights tile [n][k] (swizzled)

    constexpr int KK = KF * KF;

    const int tid  = threadIdx.x;
    const int lane = tid & 31;
    const int wid  = tid >> 5;      // 0..7
    const int wm   = wid & 3;       // 4 waves along M (32 rows each)
    const int wn   = wid >> 2;      // 2 waves along N (64 cols each)

    const int HoWo = p.Ho * p.Wo;
    const int HiWi = p.Hi * p.Wi;
    const int Kdim = p.Ci * KK;
    const int Mdim = p.N * HoWo;
    const int m0   = blockIdx.x * BM;
    const int n0   = blockIdx.y * BN;

    // ---- per-thread staging state, hoisted out of the K loop --------------
    // A: thread owns row mi (= tid&127) and K-segment ksegA (wave-uniform).
    const int mi    = tid & 127;
    const int ksegA = (tid >> 7) * 32;          // 0 or 32
    const int mA    = m0 + mi;
    const bool mvalid = mA < Mdim;
    int nimgA = mA / HoWo;
    int rA    = mA - nimgA * HoWo;
    int hoA   = rA / p.Wo;
    int woA   = rA - hoA * p.Wo;
    const int hi0   = hoA * p.S - p.P;
    const int wi0   = woA * p.S - p.P;
    const int mbase = nimgA * p.Ci * HiWi;

    // B: thread owns weight row nB (= tid&127) and 32-column segment ksegB.
    const int nB    = tid & 127;
    const int ksegB = (tid >> 7) * 32;          // 0 or 32
    const __bf16* wrow = p.Wt + (size_t)(n0 + nB) * p.Kpad + ksegB;

    v8f acc[2][4] = {};

    const int ktc = p.Kpad / BK;
    for (int kt = 0; kt < ktc; ++kt) {
        // ================= stage A (im2col, incremental k walk) ============
        {
            int k  = kt * BK + ksegA;
            int ci = k / KK;                    // compile-time divisor
            int r2 = k - ci * KK;
            int kh = r2 / KF;                   // compile-time divisor
            int kw = r2 - kh * KF;
            int addr = mbase + ci * HiWi + (hi0 + kh) * p.Wi + (wi0 + kw);
            v8bf buf;
#pragma unroll
            for (int e = 0; e < 32; ++e) {
                __bf16 v = (__bf16)0.0f;
                const int hh = hi0 + kh;
                const int ww = wi0 + kw;
                if (mvalid && k < Kdim &&
                    (unsigned)hh < (unsigned)p.Hi && (unsigned)ww < (unsigned)p.Wi)
                    v = p.X[addr];
                buf[e & 7] = v;
                if ((e & 7) == 7)
                    *(v8bf*)&lA[swz(mi, ksegA + (e - 7))] = buf;
                // advance (wave-uniform wrap logic, constant compares)
                ++k; ++addr;
                if (++kw == KF) {
                    kw = 0; addr += p.Wi - KF;
                    if (++kh == KF) { kh = 0; ++ci; addr += HiWi - KF * p.Wi; }
                }
            }
        }
        // ================= stage B (padded weights, async -> LDS) ==========
        {
            const __bf16* src = wrow + kt * BK;
#pragma unroll
            for (int c = 0; c < 32; c += 8) {
#if defined(HAVE_ASYNC_LDS)
                __builtin_amdgcn_global_load_async_to_lds_b128(
                    (v4i*)(uintptr_t)(src + c),
                    (AS3 v4i*)(unsigned)(uintptr_t)&lB[swz(nB, ksegB + c)],
                    0, 0);
#else
                *(v8bf*)&lB[swz(nB, ksegB + c)] = *(const v8bf*)(src + c);
#endif
            }
            __builtin_prefetch(src + BK, 0, 1);   // next K-tile of weights
        }
#if defined(HAVE_ASYNC_LDS)
        asm volatile("s_wait_asynccnt 0" ::: "memory");
#endif
        __syncthreads();

        // ================= compute: 2 k-steps x 2x4 WMMA ===================
        const int half = lane >> 4;
        const int lr   = lane & 15;
#pragma unroll
        for (int ks = 0; ks < 2; ++ks) {
            v16bf afrag[2], bfrag[4];
#pragma unroll
            for (int i = 0; i < 2; ++i) {
                const int row = wm * 32 + i * 16 + lr;
                v8bf lo = *(const v8bf*)&lA[swz(row, ks * 32 + half * 8)];
                v8bf hi = *(const v8bf*)&lA[swz(row, ks * 32 + 16 + half * 8)];
                afrag[i] = __builtin_shufflevector(lo, hi,
                    0,1,2,3,4,5,6,7,8,9,10,11,12,13,14,15);
            }
#pragma unroll
            for (int j = 0; j < 4; ++j) {
                const int row = wn * 64 + j * 16 + lr;
                v8bf lo = *(const v8bf*)&lB[swz(row, ks * 32 + half * 8)];
                v8bf hi = *(const v8bf*)&lB[swz(row, ks * 32 + 16 + half * 8)];
                bfrag[j] = __builtin_shufflevector(lo, hi,
                    0,1,2,3,4,5,6,7,8,9,10,11,12,13,14,15);
            }
#pragma unroll
            for (int i = 0; i < 2; ++i)
#pragma unroll
                for (int j = 0; j < 4; ++j)
                    acc[i][j] = __builtin_amdgcn_wmma_f32_16x16x32_bf16(
                        false, afrag[i], false, bfrag[j],
                        (short)0, acc[i][j], false, false);
        }
        __syncthreads();
    }

    // ------------- epilogue: BN affine + ReLU + dual store (incremental) ---
    const int halfe = lane >> 4;
    const int lre   = lane & 15;
    int nimg0[2], ho0[2], wo0[2], mst[2];
#pragma unroll
    for (int i = 0; i < 2; ++i) {
        const int ms = m0 + wm * 32 + i * 16 + halfe * 8;
        mst[i] = ms;
        const int ni = ms / HoWo;
        const int r  = ms - ni * HoWo;
        const int ho = r / p.Wo;
        nimg0[i] = ni; ho0[i] = ho; wo0[i] = r - ho * p.Wo;
    }
#pragma unroll
    for (int j = 0; j < 4; ++j) {
        const int co = n0 + wn * 64 + j * 16 + lre;     // N = lane%16
        if (co >= p.Co) continue;
        const float sc = p.gamma[co] * rsqrtf(p.var[co] + EPSV);
        const float bi = p.beta[co] - p.mean[co] * sc;
        const int coo = co + p.co_off;
#pragma unroll
        for (int i = 0; i < 2; ++i) {
            int nimg = nimg0[i], ho = ho0[i], wo = wo0[i], m = mst[i];
#pragma unroll
            for (int v = 0; v < 8; ++v) {               // M = 8*(lane/16)+v
                if (m < Mdim) {
                    float val = acc[i][j][v] * sc + bi;
                    val = val > 0.f ? val : 0.f;
                    const long idx = (((long)nimg * p.Cs + coo) * p.Ho + ho) * p.Wo + wo;
                    if (p.outF) p.outF[idx] = val;
                    if (p.outB) p.outB[idx] = (__bf16)val;
                }
                ++m;
                if (++wo == p.Wo) { wo = 0; if (++ho == p.Ho) { ho = 0; ++nimg; } }
            }
        }
    }
}

// ------------------------- correlation (441 disps, C=256) -------------------
#define CORR_C 256
#define CORR_P 21
#define CORR_D 441
__global__ __launch_bounds__(256)
void correlate_k(const float* __restrict__ f1, const float* __restrict__ f2,
                 __bf16* __restrict__ out, int N, int H, int W, int Cs)
{
    __shared__ float v1[CORR_C];
    const int HW  = H * W;
    const int pix = blockIdx.x;              // n*H*W + h*W + w
    const int n = pix / HW;
    const int r = pix - n * HW;
    const int h = r / W;
    const int w = r - h * W;

    const float* f1p = f1 + (long)n * CORR_C * HW + r;
    for (int c = threadIdx.x; c < CORR_C; c += 256)
        v1[c] = f1p[(long)c * HW];
    __syncthreads();

    const long nbase = (long)n * CORR_C * HW;
    for (int d = threadIdx.x; d < CORR_D; d += 256) {
        const int di = d / CORR_P;
        const int dj = d - di * CORR_P;
        const int hh = h + di * 2 - 20;
        const int ww = w + dj * 2 - 20;
        float s = 0.f;
        if ((unsigned)hh < (unsigned)H && (unsigned)ww < (unsigned)W) {
            const float* f2p = f2 + nbase + (long)hh * W + ww;
#pragma unroll 4
            for (int c = 0; c < CORR_C; ++c)
                s += v1[c] * f2p[(long)c * HW];
        }
        float rr = s * (1.0f / CORR_C);
        rr = rr >= 0.f ? rr : 0.1f * rr;
        out[(((long)n * Cs + d) * H + h) * W + w] = (__bf16)rr;
    }
}

// ------------------------- conversion helpers -------------------------------
// fp32 OIHW weights -> zero-padded bf16 [Copad][Kpad]
__global__ void cvt_wpad_k(const float* __restrict__ in, __bf16* __restrict__ out,
                           int Co, int Kdim, int Copad, int Kpad)
{
    const long total = (long)Copad * Kpad;
    for (long i = blockIdx.x * 256l + threadIdx.x; i < total; i += (long)gridDim.x * 256) {
        const int k  = (int)(i % Kpad);
        const int co = (int)(i / Kpad);
        const float v = (co < Co && k < Kdim) ? in[(long)co * Kdim + k] : 0.f;
        out[i] = (__bf16)v;
    }
}

// out[n][c][hw] = x[n][c0+c][hw]  (channel slice + bf16 cast)
__global__ void slice_bf16_k(const float* __restrict__ x, __bf16* __restrict__ out,
                             int N, int Ct, int c0, int C, long HW)
{
    const long total = (long)N * C * HW;
    for (long i = blockIdx.x * 256l + threadIdx.x; i < total; i += (long)gridDim.x * 256) {
        const long hw = i % HW;
        const long t  = i / HW;
        const int c = (int)(t % C);
        const int n = (int)(t / C);
        out[i] = (__bf16)x[((long)n * Ct + c0 + c) * HW + hw];
    }
}

// ---------------------------------------------------------------------------
extern "C" void kernel_launch(void* const* d_in, const int* in_sizes, int n_in,
                              void* d_out, int out_size, void* d_ws, size_t ws_size,
                              hipStream_t stream)
{
    (void)in_sizes; (void)n_in; (void)out_size; (void)ws_size;
    const float* x = (const float*)d_in[0];

    // layer table in setup_inputs dict order
    static const int CoA[11] = {64,128,256, 32,256,512,512,512,512,1024,1024};
    static const int CiA[11] = { 3, 64,128,256,473,256,512,512,512, 512,1024};
    static const int KA [11] = { 7,  5,  5,  1,  3,  3,  3,  3,  3,   3,   3};

    const float* Wf[11]; const float* Ga[11]; const float* Be[11];
    const float* Me[11]; const float* Va[11];
    for (int i = 0; i < 11; ++i) {
        Wf[i] = (const float*)d_in[1 + 5*i + 0];
        Ga[i] = (const float*)d_in[1 + 5*i + 1];
        Be[i] = (const float*)d_in[1 + 5*i + 2];
        Me[i] = (const float*)d_in[1 + 5*i + 3];
        Va[i] = (const float*)d_in[1 + 5*i + 4];
    }

    // ---- workspace bump allocator (256B aligned) --------------------------
    char* wsp = (char*)d_ws;
    size_t off = 0;
    auto alloc = [&](size_t bytes) -> void* {
        void* p = wsp + off;
        off = (off + bytes + 255) & ~(size_t)255;
        return p;
    };

    // zero-padded bf16 weights [Copad][Kpad]
    __bf16* wbf[11]; int KpadA[11], CopadA[11];
    for (int i = 0; i < 11; ++i) {
        const int Kdim = CiA[i] * KA[i] * KA[i];
        KpadA[i]  = ((Kdim + BK - 1) / BK) * BK;
        CopadA[i] = ((CoA[i] + BN - 1) / BN) * BN;
        wbf[i] = (__bf16*)alloc((size_t)CopadA[i] * KpadA[i] * 2);
    }

    const int NB = 8;
    const long HWin = 384l * 512, HW1 = 192l * 256, HW2 = 96l * 128,
               HW3 = 48l * 64,   HW4 = 24l * 32,   HW5 = 12l * 16, HW6 = 6l * 8;

    __bf16* xbf1   = (__bf16*)alloc((size_t)NB * 3 * HWin * 2);
    __bf16* xbf2   = (__bf16*)alloc((size_t)NB * 3 * HWin * 2);
    __bf16* c1a_bf = (__bf16*)alloc((size_t)NB * 64  * HW1 * 2);
    __bf16* c1b_bf = (__bf16*)alloc((size_t)NB * 64  * HW1 * 2);
    __bf16* c2a_bf = (__bf16*)alloc((size_t)NB * 128 * HW2 * 2);
    __bf16* c2b_bf = (__bf16*)alloc((size_t)NB * 128 * HW2 * 2);
    __bf16* c3a_bf = (__bf16*)alloc((size_t)NB * 256 * HW3 * 2);
    float*  c3b_f  = (float*) alloc((size_t)NB * 256 * HW3 * 4);
    __bf16* cat_bf = (__bf16*)alloc((size_t)NB * 473 * HW3 * 2);
    __bf16* c31_bf = (__bf16*)alloc((size_t)NB * 256 * HW3 * 2);
    __bf16* c4_bf  = (__bf16*)alloc((size_t)NB * 512 * HW4 * 2);
    __bf16* c41_bf = (__bf16*)alloc((size_t)NB * 512 * HW4 * 2);
    __bf16* c5_bf  = (__bf16*)alloc((size_t)NB * 512 * HW5 * 2);
    __bf16* c51_bf = (__bf16*)alloc((size_t)NB * 512 * HW5 * 2);
    __bf16* c6_bf  = (__bf16*)alloc((size_t)NB * 1024 * HW6 * 2);

    // output regions (fp32), concatenated in reference return order
    float* out_c1a = (float*)d_out;                       // 8*64*192*256
    float* out_c2a = out_c1a + (long)NB * 64  * HW1;      // 8*128*96*128
    float* out_c3a = out_c2a + (long)NB * 128 * HW2;      // 8*256*48*64
    float* out_c41 = out_c3a + (long)NB * 256 * HW3;      // 8*512*24*32
    float* out_c51 = out_c41 + (long)NB * 512 * HW4;      // 8*512*12*16
    float* out_c61 = out_c51 + (long)NB * 512 * HW5;      // 8*1024*6*8

    // ---- convert + pad weights to bf16 ------------------------------------
    for (int i = 0; i < 11; ++i) {
        const int Kdim = CiA[i] * KA[i] * KA[i];
        const long cnt = (long)CopadA[i] * KpadA[i];
        int blocks = (int)((cnt + 1023) / 1024);
        if (blocks > 2048) blocks = 2048;
        if (blocks < 1) blocks = 1;
        cvt_wpad_k<<<blocks, 256, 0, stream>>>(Wf[i], wbf[i], CoA[i], Kdim,
                                               CopadA[i], KpadA[i]);
    }
    // split x into the two 3-channel bf16 streams
    slice_bf16_k<<<2048, 256, 0, stream>>>(x, xbf1, NB, 6, 0, 3, HWin);
    slice_bf16_k<<<2048, 256, 0, stream>>>(x, xbf2, NB, 6, 3, 3, HWin);

    // ---- conv launcher (dispatch on compile-time filter size) --------------
    auto conv = [&](int li, const __bf16* X, int Hi, int Wi, int S, int P,
                    int Ho, int Wo, float* outF, __bf16* outB, int co_off, int Cs) {
        ConvParams cp;
        cp.X = X; cp.Wt = wbf[li];
        cp.gamma = Ga[li]; cp.beta = Be[li]; cp.mean = Me[li]; cp.var = Va[li];
        cp.outF = outF; cp.outB = outB;
        cp.N = NB; cp.Ci = CiA[li]; cp.Hi = Hi; cp.Wi = Wi; cp.Co = CoA[li];
        cp.Ho = Ho; cp.Wo = Wo; cp.S = S; cp.P = P;
        cp.Kpad = KpadA[li]; cp.co_off = co_off; cp.Cs = Cs;
        const long M = (long)NB * Ho * Wo;
        dim3 grid((unsigned)((M + BM - 1) / BM), (unsigned)(CopadA[li] / BN));
        switch (KA[li]) {
            case 7: conv_bn_relu_wmma<7><<<grid, dim3(256), 0, stream>>>(cp); break;
            case 5: conv_bn_relu_wmma<5><<<grid, dim3(256), 0, stream>>>(cp); break;
            case 3: conv_bn_relu_wmma<3><<<grid, dim3(256), 0, stream>>>(cp); break;
            default: conv_bn_relu_wmma<1><<<grid, dim3(256), 0, stream>>>(cp); break;
        }
    };

    // feature pyramid, both streams
    conv(0, xbf1,   384, 512, 2, 3, 192, 256, out_c1a, c1a_bf, 0, 64);
    conv(0, xbf2,   384, 512, 2, 3, 192, 256, nullptr, c1b_bf, 0, 64);
    conv(1, c1a_bf, 192, 256, 2, 2,  96, 128, out_c2a, c2a_bf, 0, 128);
    conv(1, c1b_bf, 192, 256, 2, 2,  96, 128, nullptr, c2b_bf, 0, 128);
    conv(2, c2a_bf,  96, 128, 2, 2,  48,  64, out_c3a, c3a_bf, 0, 256);
    conv(2, c2b_bf,  96, 128, 2, 2,  48,  64, c3b_f,   nullptr, 0, 256);

    // redir -> channels [441,473) of concat buffer; correlation -> [0,441)
    conv(3, c3a_bf,  48,  64, 1, 0,  48,  64, nullptr, cat_bf, 441, 473);
    correlate_k<<<(unsigned)(NB * HW3), 256, 0, stream>>>(
        out_c3a, c3b_f, cat_bf, NB, 48, 64, 473);

    // trunk
    conv(4,  cat_bf, 48, 64, 1, 1, 48, 64, nullptr, c31_bf, 0, 256);
    conv(5,  c31_bf, 48, 64, 2, 1, 24, 32, nullptr, c4_bf,  0, 512);
    conv(6,  c4_bf,  24, 32, 1, 1, 24, 32, out_c41, c41_bf, 0, 512);
    conv(7,  c41_bf, 24, 32, 2, 1, 12, 16, nullptr, c5_bf,  0, 512);
    conv(8,  c5_bf,  12, 16, 1, 1, 12, 16, out_c51, c51_bf, 0, 512);
    conv(9,  c51_bf, 12, 16, 2, 1,  6,  8, nullptr, c6_bf,  0, 1024);
    conv(10, c6_bf,   6,  8, 1, 1,  6,  8, out_c61, nullptr, 0, 1024);
}
